// Capsule_50654844289201
// MI455X (gfx1250) — matrix operand order repaired and checked by reading, not compiled, not run on hardware
//
#include <hip/hip_runtime.h>
#include <hip/hip_bf16.h>

// Capsule dynamic routing for MI455X (gfx1250), compile-only tuned.
//
// Strategy (roofline): never materialize uhat (512 MB f32 -> ~3 GB HBM traffic
// ~130us). Instead convert W once to bf16 (two layouts, L2-resident) and
// recompute the u*W contraction inside every routing pass with
// v_wmma_f32_16x16x32_bf16. Total matrix work = 5 passes * 4.3 GF = 21.5 GF,
// tiny against the bf16 WMMA ceiling; per-pass working set (64 MB W copy +
// u 4MB + logits/c 16MB) fits the 192 MB global L2.
//
// Round-4 refinements (from disasm feedback):
//  * k_agree: the per-o divergent logits RMW (s_and_saveexec + b32 load/store
//    every iteration) is replaced by 16 register-buffered partials (oo loop
//    fully unrolled) and ONE vectorized RMW per n -- half 0 lanes write o 0..7,
//    half 1 lanes write o 8..15, so no exec manipulation in the hot loop and
//    load/WMMA streams of adjacent tiles can overlap.
//  * cross-half combine uses v_permlanex16 (pure VALU, identity lane selects)
//    instead of __shfl_xor's ds_bpermute + s_wait_dscnt round-trip.

#define N_OUT 32
#define D_OUT 32
#define N_IN  2048
#define D_IN  16
#define BATCH 64
#define J_DIM (N_OUT * D_OUT)   // 1024
#define NPER  4                 // n's per workgroup in the agreement pass

typedef __attribute__((ext_vector_type(16))) __bf16 v16bf;
typedef __attribute__((ext_vector_type(8)))  __bf16 v8bf;
typedef __attribute__((ext_vector_type(8)))  float  v8f;

// cross-half (lane ^ 16) exchange as a pure-VALU permlane op
static __device__ __forceinline__ float xhalf_swap(float x) {
    int v = __builtin_amdgcn_permlanex16(__float_as_int(x), __float_as_int(x),
                                         0x76543210, 0xfedcba98, false, false);
    return __int_as_float(v);
}

// ---------------------------------------------------------------------------
// W[n][i][j] f32 -> Wo[n][i][j] bf16 (agree A-frags: 16 contiguous k) and
//                   Wt[n][j][i] bf16 (s-pass B-frags: 16 contiguous i).
// Threads map (n,j), loop i: reads and Wo writes coalesced across j.
__global__ __launch_bounds__(256) void k_convert_w(const float* __restrict__ W,
                                                   unsigned short* __restrict__ Wt_,
                                                   unsigned short* __restrict__ Wo_) {
    __bf16* Wt = (__bf16*)Wt_;
    __bf16* Wo = (__bf16*)Wo_;
    int id = blockIdx.x * blockDim.x + threadIdx.x;      // n*1024 + j
    if (id >= N_IN * J_DIM) return;
    const float* src = W + (size_t)(id >> 10) * D_IN * J_DIM + (id & 1023);
    __bf16* dst_t = Wt + (size_t)id * D_IN;
    __bf16* dst_o = Wo + (size_t)id;                     // same flat [n][i][j] index
#pragma unroll
    for (int i = 0; i < D_IN; ++i) {
        __bf16 vbf = (__bf16)src[(size_t)i * J_DIM];
        dst_t[i] = vbf;
        dst_o[(size_t)i * J_DIM] = vbf;
    }
}

// u f32 -> bf16, natural [b][n][i] layout (A fragments read 8 contiguous bf16).
__global__ __launch_bounds__(256) void k_convert_u(const float* __restrict__ u,
                                                   unsigned short* __restrict__ ub_) {
    __bf16* ub = (__bf16*)ub_;
    int id = blockIdx.x * blockDim.x + threadIdx.x;
    if (id >= BATCH * N_IN * D_IN) return;
    ub[id] = (__bf16)u[id];
}

// ---------------------------------------------------------------------------
// s-pass: s[b, j=o*32+k] += sum_{n,i} c[b,n,o]*u[b,n,i]*Wt[n][j][i]
// Grid: (o=32, split=16). 8 waves = 4 M-tiles(b) x 2 j-tiles. K-chunk = 2 n's
// (K=32), unrolled to 4 n's with dual accumulators. Split-K partials combined
// with global atomicAdd into a zeroed s.
template <bool UNIFORM>
__global__ __launch_bounds__(256) void k_s(const unsigned short* __restrict__ ub_,
                                           const unsigned short* __restrict__ Wt_,
                                           const float* __restrict__ c,
                                           float* __restrict__ s) {
    const __bf16* ub = (const __bf16*)ub_;
    const __bf16* Wt = (const __bf16*)Wt_;
    const int o     = blockIdx.x;
    const int split = blockIdx.y;
    const int w    = threadIdx.x >> 5;
    const int lane = threadIdx.x & 31;
    const int half = lane >> 4, lrow = lane & 15;
    const int m = w & 3, t = w >> 2;
    const int j    = o * D_OUT + t * 16 + lrow;   // B column / D column
    const int brow = m * 16 + lrow;               // A row (b)

    const __bf16* wt_base = Wt + (size_t)j * D_IN + half * 8; // + n0*J_DIM*D_IN
    const __bf16* u_base  = ub + (size_t)brow * N_IN * D_IN + half * 8;
    const float*  c_base  = UNIFORM ? nullptr : (c + (size_t)brow * N_IN * N_OUT + o);

    v8f acc0 = {}, acc1 = {};
    const int n_begin = split * 128;
    for (int n0 = n_begin; n0 < n_begin + 128; n0 += 4) {
        v16bf A0, A1, B0, B1;
#pragma unroll
        for (int q = 0; q < 2; ++q) {             // q-th pair of n's
            const int np = n0 + 2 * q;
            // A: elems 0..7 = (np, i=half*8..), elems 8..15 = (np+1, same i)
            const __bf16* up = u_base + (size_t)np * D_IN;
            v8bf a0 = *(const v8bf*)up;
            v8bf a1 = *(const v8bf*)(up + D_IN);
            v16bf A;
            if (UNIFORM) {
#pragma unroll
                for (int e = 0; e < 8; ++e) { A[e] = a0[e]; A[8 + e] = a1[e]; }
            } else {
                float c0 = c_base[(size_t)np * N_OUT];
                float c1 = c_base[(size_t)(np + 1) * N_OUT];
#pragma unroll
                for (int e = 0; e < 8; ++e) {
                    A[e]     = (__bf16)(c0 * (float)a0[e]);
                    A[8 + e] = (__bf16)(c1 * (float)a1[e]);
                }
            }
            // B: elems 0..7 = (np, i=half*8..), elems 8..15 = (np+1, same i)
            const __bf16* wp = wt_base + (size_t)np * J_DIM * D_IN;
            v8bf b0 = *(const v8bf*)wp;
            v8bf b1 = *(const v8bf*)(wp + (size_t)J_DIM * D_IN);
            v16bf B;
#pragma unroll
            for (int e = 0; e < 8; ++e) { B[e] = b0[e]; B[8 + e] = b1[e]; }
            if (q == 0) { A0 = A; B0 = B; } else { A1 = A; B1 = B; }
        }
        acc0 = __builtin_amdgcn_wmma_f32_16x16x32_bf16(false, A0, false, B0,
                                                       (short)0, acc0, false, false);
        acc1 = __builtin_amdgcn_wmma_f32_16x16x32_bf16(false, A1, false, B1,
                                                       (short)0, acc1, false, false);
    }
    const float scale = UNIFORM ? (1.0f / N_OUT) : 1.0f;  // softmax(0) = 1/32
#pragma unroll
    for (int p = 0; p < 8; ++p) {                 // D: lane holds rows p(+8)
        int b = m * 16 + half * 8 + p;
        atomicAdd(&s[(size_t)b * J_DIM + j], (acc0[p] + acc1[p]) * scale);
    }
}

// ---------------------------------------------------------------------------
// squash: v = (|s|^2/(1+|s|^2)) * s/|s| per (b,o) row; store bf16 v[b][j] for
// the agreement pass B-fragments; final round also writes d_out[b][o][k] f32.
__global__ __launch_bounds__(256) void k_squash(const float* __restrict__ s,
                                                unsigned short* __restrict__ vb_,
                                                float* __restrict__ out,
                                                int write_out) {
    __bf16* vb = (__bf16*)vb_;
    int id = blockIdx.x * blockDim.x + threadIdx.x;  // b*32 + o
    if (id >= BATCH * N_OUT) return;
    int b = id >> 5, o = id & 31;
    const float* row = s + (size_t)b * J_DIM + o * D_OUT;
    float n2 = 0.f;
#pragma unroll
    for (int k = 0; k < D_OUT; ++k) n2 += row[k] * row[k];
    float scale = (n2 / (1.0f + n2)) * rsqrtf(fmaxf(n2, 1e-30f));
#pragma unroll
    for (int k = 0; k < D_OUT; ++k) {
        float v = scale * row[k];
        int j = o * D_OUT + k;
        vb[(size_t)b * J_DIM + j] = (__bf16)v;
        if (write_out) out[(size_t)b * J_DIM + j] = v;
    }
}

// ---------------------------------------------------------------------------
// agreement: logits[b,n,o] += sum_{i,k} u[b,n,i]*W[n,i,o,k]*v[b,o,k]
// Per (n,o,b-tile): one full-K WMMA computes m[i,b] = sum_k W[n,i,o,k]*v[b,o,k]
//   A (16i x 32k) = Wo[n][i][o*32+k]  (two contiguous 16-byte loads/lane)
//   B (32k x 16b) = vb[b][o*32+k]     (one contiguous 32-byte load/lane)
// D tile: lane column = b, rows = i-half; lane l and l+16 hold the two
// i-halves of the SAME b, so a[b,n,o] = 8 in-register FMAs vs contiguous u
// plus one permlanex16. The 16 o-results per (b,n) are buffered in registers
// (fully unrolled oo loop) and written with a single vectorized RMW per n:
// half 0 lanes own o 0..7 of their o-half, half 1 lanes own o 8..15.
__global__ __launch_bounds__(256) void k_agree(const unsigned short* __restrict__ ub_,
                                               const unsigned short* __restrict__ Wo_,
                                               const unsigned short* __restrict__ vb_,
                                               float* __restrict__ logits) {
    const __bf16* ub = (const __bf16*)ub_;
    const __bf16* Wo = (const __bf16*)Wo_;
    const __bf16* vb = (const __bf16*)vb_;
    const int w    = threadIdx.x >> 5;
    const int lane = threadIdx.x & 31;
    const int half = lane >> 4, lrow = lane & 15;
    const int btile = w & 3, h = w >> 2;          // 4 b-tiles x 2 o-halves
    const int b = btile * 16 + lrow;              // this lane's b column
    const int nbase = blockIdx.x * NPER;

    for (int nn = nbase; nn < nbase + NPER; ++nn) {
        // u[b][nn][i], i = half*8..half*8+7: epilogue dot operand (8 contiguous)
        v8bf uu = *(const v8bf*)(ub + ((size_t)b * N_IN + nn) * D_IN + half * 8);
        const __bf16* wrow = Wo + ((size_t)nn * D_IN + lrow) * J_DIM; // A row i=lrow
        const __bf16* vrow = vb + (size_t)b * J_DIM + half * 16;      // B col b
        float vals[16];                           // a[b,nn,o] for o = h*16+oo
#pragma unroll
        for (int oo = 0; oo < 16; ++oo) {
            const int o = h * 16 + oo;
            // A-frag: row M=i=lrow; elems 0..15 = k(half*16..half*16+15)
            const __bf16* ap = wrow + o * D_OUT + half * 16;
            v8bf alo = *(const v8bf*)ap;
            v8bf ahi = *(const v8bf*)(ap + 8);
            v16bf A;
#pragma unroll
            for (int e = 0; e < 8; ++e) { A[e] = alo[e]; A[8 + e] = ahi[e]; }
            // B-frag: col N=lrow (b); K rows half*16..half*16+15 (contiguous k)
            v16bf B = *(const v16bf*)(vrow + (size_t)o * D_OUT);
            v8f cz = {};
            v8f D = __builtin_amdgcn_wmma_f32_16x16x32_bf16(false, A, false, B,
                                                            (short)0, cz, false, false);
            // a-partial over this lane's i-half, then combine halves (VALU)
            float partial = 0.f;
#pragma unroll
            for (int p = 0; p < 8; ++p) partial += D[p] * (float)uu[p];
            vals[oo] = partial + xhalf_swap(partial);
        }
        // one vectorized RMW per n: 8 consecutive o's per lane, all lanes active
        float* lp = logits + ((size_t)b * N_IN + nn) * N_OUT + h * 16 + half * 8;
        v8f oldv = *(v8f*)lp;
#pragma unroll
        for (int p = 0; p < 8; ++p) oldv[p] += half ? vals[8 + p] : vals[p];
        *(v8f*)lp = oldv;
    }
}

// ---------------------------------------------------------------------------
__global__ __launch_bounds__(256) void k_softmax(const float* __restrict__ logits,
                                                 float* __restrict__ c) {
    int id = blockIdx.x * blockDim.x + threadIdx.x;  // b*N_IN + n
    if (id >= BATCH * N_IN) return;
    const float* row = logits + (size_t)id * N_OUT;
    float mx = row[0];
#pragma unroll
    for (int o = 1; o < N_OUT; ++o) mx = fmaxf(mx, row[o]);
    float e[N_OUT], sum = 0.f;
#pragma unroll
    for (int o = 0; o < N_OUT; ++o) { e[o] = __expf(row[o] - mx); sum += e[o]; }
    float inv = 1.0f / sum;
    float* crow = c + (size_t)id * N_OUT;
#pragma unroll
    for (int o = 0; o < N_OUT; ++o) crow[o] = e[o] * inv;
}

// ---------------------------------------------------------------------------
extern "C" void kernel_launch(void* const* d_in, const int* in_sizes, int n_in,
                              void* d_out, int out_size, void* d_ws, size_t ws_size,
                              hipStream_t stream) {
    const float* u = (const float*)d_in[0];   // [64][2048][16]
    const float* W = (const float*)d_in[1];   // [2048][16][1024]
    float* out = (float*)d_out;               // [64][32][32]

    char* ws = (char*)d_ws;
    size_t off = 0;
    auto bump = [&](size_t bytes) {
        void* p = (void*)(ws + off);
        off += (bytes + 255) & ~(size_t)255;
        return p;
    };
    unsigned short* Wt  = (unsigned short*)bump((size_t)N_IN * J_DIM * D_IN * 2);  // 64 MB bf16 [n][j][i]
    unsigned short* Wo  = (unsigned short*)bump((size_t)N_IN * D_IN * J_DIM * 2);  // 64 MB bf16 [n][i][j]
    unsigned short* ubf = (unsigned short*)bump((size_t)BATCH * N_IN * D_IN * 2);  // 4 MB bf16
    float* logits = (float*)bump((size_t)BATCH * N_IN * N_OUT * 4);                // 16 MB
    float* cbuf   = (float*)bump((size_t)BATCH * N_IN * N_OUT * 4);                // 16 MB
    float* sbuf   = (float*)bump((size_t)BATCH * J_DIM * 4);                       // 256 KB
    unsigned short* vbf = (unsigned short*)bump((size_t)BATCH * J_DIM * 2);        // 128 KB bf16
    const size_t S_BYTES = (size_t)BATCH * J_DIM * 4;
    const size_t L_BYTES = (size_t)BATCH * N_IN * N_OUT * 4;

    // One-time (per call) conversion + logits init
    k_convert_w<<<(N_IN * J_DIM) / 256, 256, 0, stream>>>(W, Wt, Wo);
    k_convert_u<<<(BATCH * N_IN * D_IN) / 256, 256, 0, stream>>>(u, ubf);
    hipMemsetAsync(logits, 0, L_BYTES, stream);

    // round 0: c uniform (softmax of zeros), folded as a 1/32 epilogue scale
    hipMemsetAsync(sbuf, 0, S_BYTES, stream);
    k_s<true><<<dim3(N_OUT, 16), 256, 0, stream>>>(ubf, Wt, (const float*)nullptr, sbuf);
    k_squash<<<(BATCH * N_OUT) / 256, 256, 0, stream>>>(sbuf, vbf, out, 0);

    // rounds 1..2: agreement -> softmax -> s -> squash
    for (int r = 1; r < 3; ++r) {
        k_agree<<<N_IN / NPER, 256, 0, stream>>>(ubf, Wo, vbf, logits);
        k_softmax<<<(BATCH * N_IN) / 256, 256, 0, stream>>>(logits, cbuf);
        hipMemsetAsync(sbuf, 0, S_BYTES, stream);
        k_s<false><<<dim3(N_OUT, 16), 256, 0, stream>>>(ubf, Wt, cbuf, sbuf);
        k_squash<<<(BATCH * N_OUT) / 256, 256, 0, stream>>>(sbuf, vbf, out, r == 2 ? 1 : 0);
    }
}
